// MultiheadAttention_9775345566095
// MI455X (gfx1250) — matrix operand order repaired and checked by reading, not compiled, not run on hardware
//
#include <hip/hip_runtime.h>
#include <hip/hip_bf16.h>

// ---------------------------------------------------------------------------
// MI455X (gfx1250, wave32) multi-head causal attention, bf16 WMMA pipeline.
//   x[8192,1024] fp32, W* [1024,1024] fp32 (torch-style [out,in]).
//   QKV proj -> flash attention (TDM-fed tiles) -> output proj,
//   all matmuls on v_wmma_f32_16x16x32_bf16.
// ---------------------------------------------------------------------------

typedef __attribute__((ext_vector_type(16))) __bf16          v16bf;
typedef __attribute__((ext_vector_type(8)))  float           v8f;
typedef __attribute__((ext_vector_type(8)))  unsigned short  ushort8;
typedef __attribute__((ext_vector_type(16))) unsigned short  ushort16;
typedef __attribute__((ext_vector_type(4)))  unsigned int    uint4v;
typedef __attribute__((ext_vector_type(8)))  unsigned int    uint8v;

static __device__ __forceinline__ unsigned short f2bf(float f) {
    union { float f; unsigned u; } v; v.f = f;
    unsigned r = v.u + 0x7FFFu + ((v.u >> 16) & 1u);   // round-to-nearest-even
    return (unsigned short)(r >> 16);
}
static __device__ __forceinline__ float bf2f(unsigned short h) {
    union { unsigned u; float f; } v; v.u = ((unsigned)h) << 16;
    return v.f;
}
static __device__ __forceinline__ unsigned short to_bf16(float f)          { return f2bf(f); }
static __device__ __forceinline__ unsigned short to_bf16(unsigned short u) { return u; }

static __device__ __forceinline__ v16bf join16(ushort8 lo, ushort8 hi) {
    union { ushort16 u; v16bf b; } x;
#pragma unroll
    for (int i = 0; i < 8; ++i) { x.u[i] = lo[i]; x.u[8 + i] = hi[i]; }
    return x.b;
}
static __device__ __forceinline__ void store_out(float* p, float v)          { *p = v; }
static __device__ __forceinline__ void store_out(unsigned short* p, float v) { *p = f2bf(v); }

// ---------------------------------------------------------------------------
// Tensor Data Mover: DMA a 2-D bf16 tile [d1 rows x d0 elems] from global
// (row stride = stride_elems) into LDS at lds_off, padding each 128 B row
// with 16 B so rows land on a 144 B (72-ushort) stride.
// D# group0: count=1 | lds_addr | global_addr[56:0] | type=2.
// D# group1: data_size=2B, pad_enable, pad_interval=32 DW, pad_amount=4 DW,
//            tensor/tile dims + dim0 stride. Groups 2/3 zero (2-D tensor).
// ---------------------------------------------------------------------------
static __device__ __forceinline__ void tdm_load_tile_bf16(
        unsigned lds_off, const unsigned short* gaddr,
        unsigned d0, unsigned d1, unsigned stride_elems) {
    unsigned long long ga = (unsigned long long)(const void*)gaddr;
    uint4v g0;
    g0[0] = 1u;                                            // count=1 (valid D#)
    g0[1] = lds_off;                                       // lds_addr (bytes)
    g0[2] = (unsigned)(ga & 0xFFFFFFFFu);                  // global_addr[31:0]
    g0[3] = (unsigned)((ga >> 32) & 0x01FFFFFFu)           // global_addr[56:32]
          | (2u << 30);                                    // type = 2 (image)
    uint8v g1;
    g1[0] = (1u << 16)                                     // data_size = 2 bytes
          | (1u << 20)                                     // pad_enable
          | (4u << 22)                                     // pad_interval: 32 DWORDs (128 B)
          | (3u << 25);                                    // pad_amount:   4 DWORDs (16 B)
    g1[1] = (d0 & 0xFFFFu) << 16;                          // tensor_dim0[15:0]
    g1[2] = (d1 & 0xFFFFu) << 16;                          // tensor_dim1[15:0] (dim0 hi = 0)
    g1[3] = (d0 & 0xFFFFu) << 16;                          // tile_dim0 (dim1 hi = 0)
    g1[4] = (d1 & 0xFFFFu);                                // tile_dim1 (tile_dim2 = 0)
    g1[5] = stride_elems;                                  // tensor_dim0_stride[31:0]
    g1[6] = 0u;                                            // stride hi / dim1_stride lo
    g1[7] = 0u;                                            // dim1_stride hi
    uint4v gz; gz[0] = 0u; gz[1] = 0u; gz[2] = 0u; gz[3] = 0u;
    asm volatile("tensor_load_to_lds %0, %1, %2, %3"
                 :: "s"(g0), "s"(g1), "s"(gz), "s"(gz)
                 : "memory");
}

// ---------------------------------------------------------------------------
// GEMM: C[M,N] = A[M,K] * W[N,K]^T   (A fp32 or bf16-ushort; W fp32; C fp32 or bf16)
// Block: 256 threads = 8 waves; 128x128 tile; K-step 32 (one bf16 WMMA K).
// Wave (wr=wave&3, wc=wave>>2) owns a 32x64 sub-tile = 2x4 WMMA accumulators.
// (fp32 -> bf16 conversion must pass through the VALU, so tiles are staged
//  by the waves rather than the TDM here.)
// ---------------------------------------------------------------------------
template <typename InT, typename OutT>
__global__ __launch_bounds__(256)
void gemm_xwT_bf16(const InT* __restrict__ A, const float* __restrict__ W,
                   OutT* __restrict__ C, int M, int N, int K) {
    __shared__ __align__(16) unsigned short As[128][40];  // 128 rows x 32 K (+pad)
    __shared__ __align__(16) unsigned short Ws[128][40];  // 128 N    x 32 K (+pad)

    const int tid  = threadIdx.x;
    const int wave = tid >> 5, lane = tid & 31;
    const int wr   = wave & 3, wc = wave >> 2;
    const int h    = lane >> 4, ln = lane & 15;

    const int m0 = blockIdx.x * 128;
    const int n0 = blockIdx.y * 128;

    const int lrow = tid >> 1;        // 0..127 (2 threads per row)
    const int lcol = (tid & 1) * 16;  // 0 or 16

    v8f acc[2][4];
    v8f z = {};
#pragma unroll
    for (int i = 0; i < 2; ++i)
#pragma unroll
        for (int j = 0; j < 4; ++j) acc[i][j] = z;

    for (int k0 = 0; k0 < K; k0 += 32) {
        // ---- stage A and W tiles as bf16 into LDS ----
        {
            const InT*   ap = A + (size_t)(m0 + lrow) * K + k0 + lcol;
            const float* wp = W + (size_t)(n0 + lrow) * K + k0 + lcol;
#pragma unroll
            for (int j = 0; j < 16; ++j) As[lrow][lcol + j] = to_bf16(ap[j]);
#pragma unroll
            for (int j = 0; j < 16; ++j) Ws[lrow][lcol + j] = f2bf(wp[j]);
            if (k0 + 32 < K) {  // hint next K-step tiles toward the caches
                __builtin_prefetch(ap + 32, 0, 3);
                __builtin_prefetch(wp + 32, 0, 3);
            }
        }
        __syncthreads();

        // ---- fragments per ISA layouts ----
        v16bf a[2], b[4];
#pragma unroll
        for (int sm = 0; sm < 2; ++sm) {
            const unsigned short* p = &As[wr * 32 + sm * 16 + ln][h * 8];
            a[sm] = join16(*(const ushort8*)p, *(const ushort8*)(p + 16));
        }
#pragma unroll
        for (int sn = 0; sn < 4; ++sn) {
            const unsigned short* p = &Ws[wc * 64 + sn * 16 + ln][h * 8];
            b[sn] = join16(*(const ushort8*)p, *(const ushort8*)(p + 16));
        }
#pragma unroll
        for (int sm = 0; sm < 2; ++sm)
#pragma unroll
            for (int sn = 0; sn < 4; ++sn)
                acc[sm][sn] = __builtin_amdgcn_wmma_f32_16x16x32_bf16(
                    false, a[sm], false, b[sn], (short)0, acc[sm][sn], false, false);
        __syncthreads();
    }

    // ---- epilogue: C layout row = 8*h + v, col = ln ----
#pragma unroll
    for (int sm = 0; sm < 2; ++sm)
#pragma unroll
        for (int sn = 0; sn < 4; ++sn)
#pragma unroll
            for (int v = 0; v < 8; ++v) {
                int row = m0 + wr * 32 + sm * 16 + h * 8 + v;
                int col = n0 + wc * 64 + sn * 16 + ln;
                store_out(C + (size_t)row * N + col, acc[sm][sn][v]);
            }
}

// ---------------------------------------------------------------------------
// Flash attention (causal), bf16 Q/K/V -> bf16 O, fp32 accumulation.
// Block: 128 threads = 4 waves; 64 query rows/block (16 per wave); dk = 64.
// Q and K tiles are DMA'd into (padded) LDS by the Tensor Data Mover; V is
// transposed on the way in by the waves. Online softmax, WMMA throughout.
// ---------------------------------------------------------------------------
__global__ __launch_bounds__(128)
void flash_attn_fwd(const unsigned short* __restrict__ Q,
                    const unsigned short* __restrict__ K,
                    const unsigned short* __restrict__ V,
                    unsigned short* __restrict__ O,
                    int S, int D, int H) {
    __shared__ __align__(16) unsigned short Qs[64][72];      // [q][dk]    (TDM)
    __shared__ __align__(16) unsigned short Ks[64][72];      // [key][dk]  (TDM)
    __shared__ __align__(16) unsigned short Vt[64][72];      // [dk][key]  (transposed)
    __shared__ __align__(16) unsigned short Ps[4][16][72];   // per-wave P staging

    const int tid  = threadIdx.x;
    const int wave = tid >> 5, lane = tid & 31;
    const int h2   = lane >> 4, ln = lane & 15;
    const int b    = blockIdx.x / H;
    const int hd   = blockIdx.x % H;
    const int q0   = blockIdx.y * 64;
    const size_t rowbase = (size_t)b * S;
    const int cb   = hd * 64;

    const int lrow = tid >> 1;         // 0..63
    const int lcol = (tid & 1) * 32;   // 0 or 32

    // ---- TDM: DMA the Q tile (raw; 1/sqrt(dk) folded into scores later) ----
    if (wave == 0) {
        tdm_load_tile_bf16((unsigned)(size_t)(void*)&Qs[0][0],
                           Q + (rowbase + q0) * D + cb, 64u, 64u, (unsigned)D);
    }

    float m_i[8], l_i[8];
    v8f o_acc[4];
    v8f zf = {};
#pragma unroll
    for (int v = 0; v < 8; ++v) { m_i[v] = -1e30f; l_i[v] = 0.f; }
#pragma unroll
    for (int d = 0; d < 4; ++d) o_acc[d] = zf;

    const int ntiles = q0 / 64 + 1;   // causal: only tiles with kv0 <= q0
    for (int t = 0; t < ntiles; ++t) {
        const int kv0 = t * 64;
        __syncthreads();  // all waves done reading tiles of prior iteration

        // ---- TDM: DMA the K tile while the waves transpose V ----
        if (wave == 0) {
            tdm_load_tile_bf16((unsigned)(size_t)(void*)&Ks[0][0],
                               K + (rowbase + kv0) * D + cb, 64u, 64u, (unsigned)D);
        }
        {
            const unsigned short* vp = V + (rowbase + kv0 + lrow) * D + cb + lcol;
#pragma unroll
            for (int j = 0; j < 32; ++j)
                Vt[lcol + j][lrow] = vp[j];   // transpose V on the way in
        }
        if (wave == 0) __builtin_amdgcn_s_wait_tensorcnt(0);  // Q (t==0) + K done
        __syncthreads();

        // ---- S = Q * K^T : 4 n-tiles x 2 K-steps = 8 WMMAs/wave ----
        v8f s[4];
        s[0] = s[1] = s[2] = s[3] = zf;
        v16bf aq[2];
#pragma unroll
        for (int ks = 0; ks < 2; ++ks) {
            const unsigned short* p = &Qs[wave * 16 + ln][ks * 32 + h2 * 8];
            aq[ks] = join16(*(const ushort8*)p, *(const ushort8*)(p + 16));
        }
#pragma unroll
        for (int nt = 0; nt < 4; ++nt)
#pragma unroll
            for (int ks = 0; ks < 2; ++ks) {
                const unsigned short* p = &Ks[nt * 16 + ln][ks * 32 + h2 * 8];
                v16bf bk = join16(*(const ushort8*)p, *(const ushort8*)(p + 16));
                s[nt] = __builtin_amdgcn_wmma_f32_16x16x32_bf16(
                    false, aq[ks], false, bk, (short)0, s[nt], false, false);
            }

        // ---- scale by 1/sqrt(dk) ----
#pragma unroll
        for (int nt = 0; nt < 4; ++nt)
#pragma unroll
            for (int v = 0; v < 8; ++v) s[nt][v] *= 0.125f;

        // ---- causal mask (only needed on/after the diagonal tile) ----
        const int qrow_base = q0 + wave * 16 + h2 * 8;   // + v
        if (kv0 + 63 > q0 + wave * 16) {
#pragma unroll
            for (int nt = 0; nt < 4; ++nt) {
                int key = kv0 + nt * 16 + ln;
#pragma unroll
                for (int v = 0; v < 8; ++v)
                    if (key > qrow_base + v) s[nt][v] = -1e30f;
            }
        }

        // ---- online softmax (row stats across 16-lane halves) ----
#pragma unroll
        for (int v = 0; v < 8; ++v) {
            float mx = fmaxf(fmaxf(s[0][v], s[1][v]), fmaxf(s[2][v], s[3][v]));
#pragma unroll
            for (int off = 1; off < 16; off <<= 1)
                mx = fmaxf(mx, __shfl_xor(mx, off, 32));
            float mnew = fmaxf(m_i[v], mx);
            float corr = __expf(m_i[v] - mnew);
            float rs = 0.f;
#pragma unroll
            for (int nt = 0; nt < 4; ++nt) {
                float p = __expf(s[nt][v] - mnew);
                s[nt][v] = p;
                rs += p;
            }
#pragma unroll
            for (int off = 1; off < 16; off <<= 1)
                rs += __shfl_xor(rs, off, 32);
            l_i[v] = l_i[v] * corr + rs;
            m_i[v] = mnew;
#pragma unroll
            for (int d = 0; d < 4; ++d) o_acc[d][v] *= corr;
        }

        // ---- transpose P through wave-private LDS into A-fragment layout ----
#pragma unroll
        for (int nt = 0; nt < 4; ++nt)
#pragma unroll
            for (int v = 0; v < 8; ++v)
                Ps[wave][h2 * 8 + v][nt * 16 + ln] = f2bf(s[nt][v]);
        // same-wave LDS ops are in-order (DScnt): reads below see the writes

        // ---- O += P * V : 4 dk-tiles x 2 K-steps = 8 WMMAs/wave ----
        v16bf ap[2];
#pragma unroll
        for (int ks = 0; ks < 2; ++ks) {
            const unsigned short* p = &Ps[wave][ln][ks * 32 + h2 * 8];
            ap[ks] = join16(*(const ushort8*)p, *(const ushort8*)(p + 16));
        }
#pragma unroll
        for (int d = 0; d < 4; ++d)
#pragma unroll
            for (int ks = 0; ks < 2; ++ks) {
                const unsigned short* p = &Vt[d * 16 + ln][ks * 32 + h2 * 8];
                v16bf bv = join16(*(const ushort8*)p, *(const ushort8*)(p + 16));
                o_acc[d] = __builtin_amdgcn_wmma_f32_16x16x32_bf16(
                    false, ap[ks], false, bv, (short)0, o_acc[d], false, false);
            }
    }

    // ---- normalize and write attn output (bf16, [B*S, D] with head columns) ----
#pragma unroll
    for (int v = 0; v < 8; ++v) {
        float inv = 1.0f / l_i[v];
        size_t row = rowbase + q0 + wave * 16 + h2 * 8 + v;
#pragma unroll
        for (int d = 0; d < 4; ++d)
            O[row * D + cb + d * 16 + ln] = f2bf(o_acc[d][v] * inv);
    }
}

// ---------------------------------------------------------------------------
extern "C" void kernel_launch(void* const* d_in, const int* in_sizes, int n_in,
                              void* d_out, int out_size, void* d_ws, size_t ws_size,
                              hipStream_t stream) {
    const float* x  = (const float*)d_in[0];
    const float* wq = (const float*)d_in[1];
    const float* wk = (const float*)d_in[2];
    const float* wv = (const float*)d_in[3];
    const float* wo = (const float*)d_in[4];

    const int B = 4, S = 2048, D = 1024, H = 16;
    const int M = B * S;                       // 8192 rows
    const size_t mat = (size_t)M * D;          // elements per bf16 buffer

    unsigned short* qb = (unsigned short*)d_ws;      // Q  bf16  (16 MB)
    unsigned short* kb = qb + mat;                   // K  bf16  (16 MB)
    unsigned short* vb = kb + mat;                   // V  bf16  (16 MB)
    unsigned short* ab = vb + mat;                   // attn out bf16 (16 MB)

    dim3 ggrid(M / 128, D / 128);                    // 64 x 8 tiles
    gemm_xwT_bf16<float, unsigned short><<<ggrid, 256, 0, stream>>>(x, wq, qb, M, D, D);
    gemm_xwT_bf16<float, unsigned short><<<ggrid, 256, 0, stream>>>(x, wk, kb, M, D, D);
    gemm_xwT_bf16<float, unsigned short><<<ggrid, 256, 0, stream>>>(x, wv, vb, M, D, D);

    flash_attn_fwd<<<dim3(B * H, S / 64), 128, 0, stream>>>(qb, kb, vb, ab, S, D, H);

    gemm_xwT_bf16<unsigned short, float><<<ggrid, 256, 0, stream>>>(ab, wo, (float*)d_out, M, D, D);
}
